// GRUModel_27779848470913
// MI455X (gfx1250) — matrix operand order repaired and checked by reading, not compiled, not run on hardware
//
#include <hip/hip_runtime.h>
#include <math.h>

// ---------------- problem constants ----------------
#define T_LEN 256
#define B_SZ  2048
#define DIN   364
#define E1    128
#define H_SZ  96
#define NACT  8

typedef __attribute__((ext_vector_type(16))) _Float16 v16h;
typedef __attribute__((ext_vector_type(8)))  _Float16 v8h;
typedef __attribute__((ext_vector_type(8)))  float    v8f;
typedef __attribute__((ext_vector_type(4)))  float    v4f;

// ---------------- workspace layout (bytes) ----------------
// packed f16 B-matrices (lane-contiguous WMMA fragments), then e[T,B,96] f16
static constexpr size_t OFF_W1P  = 0;                                   // 12 kc * 8 nt * 512 halves
static constexpr size_t OFF_W2P  = OFF_W1P  + (size_t)12*8*512*2;       //  4 kc * 6 nt * 512
static constexpr size_t OFF_WIH0 = OFF_W2P  + (size_t)4*6*512*2;        //  3 kc * 18 nt * 512 each
static constexpr size_t OFF_WIH1 = OFF_WIH0 + (size_t)3*18*512*2;
static constexpr size_t OFF_WHH0 = OFF_WIH1 + (size_t)3*18*512*2;       // whh0, whh1 contiguous
static constexpr size_t OFF_WHH1 = OFF_WHH0 + (size_t)3*18*512*2;
static constexpr size_t OFF_E    = OFF_WHH1 + (size_t)3*18*512*2;       // e [T,B,96] f16

#define GRU_WMAT_H (3*18*512)          // 27648 halves per GRU weight matrix

__device__ __forceinline__ v8f wmma16(v16h a, v16h b, v8f c) {
  return __builtin_amdgcn_wmma_f32_16x16x32_f16(false, a, false, b, (short)0, c, false, false);
}

// Pack f32 weight W[Nrows, Kdim] (row-major; used as B = W^T : K x N) into WMMA
// B-fragment layout: elem j of lane l of tile (kc,nt): K = kc*32 + (l>=16?8:0) + j + (j>=8?8:0),
// N = nt*16 + (l&15); zero-pad out of range.
__global__ void pack_b_kernel(const float* __restrict__ w, _Float16* __restrict__ dst,
                              int Kdim, int Nrows, int Ntiles, int total) {
  int idx = blockIdx.x * blockDim.x + threadIdx.x;
  if (idx >= total) return;
  int j    = idx & 15;
  int lane = (idx >> 4) & 31;
  int tile = idx >> 9;
  int nt = tile % Ntiles;
  int kc = tile / Ntiles;
  int K = kc * 32 + ((lane >= 16) ? 8 : 0) + j + ((j >= 8) ? 8 : 0);
  int N = nt * 16 + (lane & 15);
  float v = (K < Kdim && N < Nrows) ? w[(size_t)N * Kdim + K] : 0.0f;
  dst[idx] = (_Float16)v;
}

// Fused embedding MLP:  e = LeakyReLU(x @ W1^T + b1) @ W2^T + b2  (16-row tile per wave)
// x is streamed once with non-temporal loads so the 763MB sweep does not evict the
// 96MB e tensor from the 192MB L2 (the GRU kernel re-reads e right after).
__global__ void __launch_bounds__(256) embed_kernel(
    const float* __restrict__ x, const _Float16* __restrict__ w1p,
    const float* __restrict__ b1, const _Float16* __restrict__ w2p,
    const float* __restrict__ b2, _Float16* __restrict__ e)
{
  __shared__ _Float16 tile[8][16][E1];           // 32 KB: per-wave 16x128 stage-1 result
  const int lane   = threadIdx.x & 31;
  const int wave   = threadIdx.x >> 5;
  const int m0     = (blockIdx.x * 8 + wave) * 16;
  const int col    = lane & 15;
  const int kb     = ((lane >> 4) & 1) * 8;      // A-fragment K base (0 or 8)
  const int rowOff = ((lane >> 4) & 1) * 8;      // C-fragment M base (0 or 8)
  const int rowA   = m0 + (lane & 15);

  v8f acc1[8];
#pragma unroll
  for (int i = 0; i < 8; ++i) acc1[i] = (v8f){};
  const float* xrow = x + (size_t)rowA * DIN;    // row stride 1456B = 91*16 -> 16B aligned
  for (int kc = 0; kc < 12; ++kc) {
    union { v16h v; _Float16 h[16]; } a;
    const int base = kc * 32 + kb;
    if (kc < 11) {                               // fully in-bounds, aligned float4 NT loads
      v4f p0 = __builtin_nontemporal_load((const v4f*)(xrow + base));
      v4f p1 = __builtin_nontemporal_load((const v4f*)(xrow + base + 4));
      v4f p2 = __builtin_nontemporal_load((const v4f*)(xrow + base + 16));
      v4f p3 = __builtin_nontemporal_load((const v4f*)(xrow + base + 20));
#pragma unroll
      for (int j = 0; j < 4; ++j) {
        a.h[j]      = (_Float16)p0[j];
        a.h[j + 4]  = (_Float16)p1[j];
        a.h[j + 8]  = (_Float16)p2[j];
        a.h[j + 12] = (_Float16)p3[j];
      }
    } else {                                     // tail chunk: guarded scalar loads
#pragma unroll
      for (int j = 0; j < 8; ++j) {
        int k0 = base + j, k1 = base + 16 + j;
        a.h[j]     = (k0 < DIN) ? (_Float16)__builtin_nontemporal_load(xrow + k0) : (_Float16)0.0f;
        a.h[j + 8] = (k1 < DIN) ? (_Float16)__builtin_nontemporal_load(xrow + k1) : (_Float16)0.0f;
      }
    }
#pragma unroll
    for (int nt = 0; nt < 8; ++nt) {
      v16h b = *(const v16h*)(w1p + (((size_t)(kc * 8 + nt)) * 32 + lane) * 16);
      acc1[nt] = wmma16(a.v, b, acc1[nt]);
    }
  }
#pragma unroll
  for (int nt = 0; nt < 8; ++nt) {
#pragma unroll
    for (int r = 0; r < 8; ++r) {
      int N = nt * 16 + col;
      float v = acc1[nt][r] + b1[N];
      v = (v > 0.0f) ? v : 0.01f * v;
      tile[wave][rowOff + r][N] = (_Float16)v;
    }
  }
  __syncthreads();

  v8f acc2[6];
#pragma unroll
  for (int i = 0; i < 6; ++i) acc2[i] = (v8f){};
  const int mA = lane & 15;
#pragma unroll
  for (int kc = 0; kc < 4; ++kc) {
    union { v16h v; v8h h[2]; } a;
    const int base = kc * 32 + kb;
    a.h[0] = *(const v8h*)&tile[wave][mA][base];
    a.h[1] = *(const v8h*)&tile[wave][mA][base + 16];
#pragma unroll
    for (int nt = 0; nt < 6; ++nt) {
      v16h b = *(const v16h*)(w2p + (((size_t)(kc * 6 + nt)) * 32 + lane) * 16);
      acc2[nt] = wmma16(a.v, b, acc2[nt]);
    }
  }
#pragma unroll
  for (int nt = 0; nt < 6; ++nt) {
#pragma unroll
    for (int r = 0; r < 8; ++r) {
      int N = nt * 16 + col;
      int row = m0 + rowOff + r;
      e[(size_t)row * H_SZ + N] = (_Float16)(acc2[nt][r] + b2[N]);
    }
  }
}

// One GRU cell update for 16 rows x 16 cols (one column-tile wave, one timestep).
// xt/hprev/hnext are LDS tiles [16][96] f16; wih streams from WGP$/L2 (loop-invariant
// addresses), whh from LDS. State hs[8] stays in f32 registers of the owning lane.
__device__ __forceinline__ void gru_cell(
    const _Float16* __restrict__ xt, const _Float16* __restrict__ hprev,
    const _Float16* __restrict__ wih, const _Float16* __restrict__ whh,
    int colTile, int lane,
    float bir, float bhr, float biz, float bhz, float bin_, float bhn,
    float* hs, const int* len, int t, _Float16* hnext)
{
  const int kb     = ((lane >> 4) & 1) * 8;
  const int mA     = lane & 15;
  const int col    = lane & 15;
  const int rowOff = ((lane >> 4) & 1) * 8;

  v8f xr = {}, hr = {}, xz = {}, hz = {}, xn = {}, hn = {};
#pragma unroll
  for (int kc = 0; kc < 3; ++kc) {
    union { v16h v; v8h h[2]; } ax, ah;
    const int base = kc * 32 + kb;
    ax.h[0] = *(const v8h*)(xt + mA * H_SZ + base);
    ax.h[1] = *(const v8h*)(xt + mA * H_SZ + base + 16);
    ah.h[0] = *(const v8h*)(hprev + mA * H_SZ + base);
    ah.h[1] = *(const v8h*)(hprev + mA * H_SZ + base + 16);
#pragma unroll
    for (int g = 0; g < 3; ++g) {
      size_t tix = ((size_t)(kc * 18 + g * 6 + colTile) * 32 + lane) * 16;
      v16h bi = *(const v16h*)(wih + tix);
      v16h bh = *(const v16h*)(whh + tix);
      if      (g == 0) { xr = wmma16(ax.v, bi, xr); hr = wmma16(ah.v, bh, hr); }
      else if (g == 1) { xz = wmma16(ax.v, bi, xz); hz = wmma16(ah.v, bh, hz); }
      else             { xn = wmma16(ax.v, bi, xn); hn = wmma16(ah.v, bh, hn); }
    }
  }
  const int N = colTile * 16 + col;
#pragma unroll
  for (int r = 0; r < 8; ++r) {
    float rr = 1.0f / (1.0f + expf(-(xr[r] + bir + hr[r] + bhr)));
    float zz = 1.0f / (1.0f + expf(-(xz[r] + biz + hz[r] + bhz)));
    float nn = tanhf(xn[r] + bin_ + rr * (hn[r] + bhn));
    float hv = (t < len[r]) ? ((1.0f - zz) * nn + zz * hs[r]) : hs[r];
    hs[r] = hv;
    hnext[(rowOff + r) * H_SZ + N] = (_Float16)hv;
  }
}

// Persistent fused 2-layer GRU: one block owns 16 batch rows for all 256 steps.
// 6 waves = 6 column tiles of H. Layer-1 input is layer-0's fresh LDS buffer
// (hs1 never touches global memory). e_{t+1} is prefetched into a register during
// step t's WMMA work so the global load latency never sits on the serial path.
__global__ void __launch_bounds__(192) gru_fused_kernel(
    const _Float16* __restrict__ e,
    const _Float16* __restrict__ wih0, const _Float16* __restrict__ wih1,
    const _Float16* __restrict__ whh01,   // whh0 then whh1, contiguous
    const float* __restrict__ b_ih0, const float* __restrict__ b_hh0,
    const float* __restrict__ b_ih1, const float* __restrict__ b_hh1,
    const float* __restrict__ h0g, const int* __restrict__ lengths,
    float* __restrict__ hout)             // [2,B,96] section of d_out
{
  __shared__ __align__(32) _Float16 whh_lds[2 * GRU_WMAT_H];  // 110,592 B
  __shared__ __align__(16) _Float16 hbuf[2][2][16][H_SZ];     //  12,288 B [layer][pp][row][col]
  __shared__ __align__(16) _Float16 elds[16 * H_SZ];          //   3,072 B

  const int tid  = threadIdx.x;
  const int lane = tid & 31;
  const int wave = tid >> 5;             // colTile 0..5
  const int m0   = blockIdx.x * 16;
  const int col  = lane & 15;
  const int rowOff = ((lane >> 4) & 1) * 8;
  const int N    = wave * 16 + col;

  // stage whh0|whh1 into LDS (6912 uint4 / 192 threads = 36 each)
  {
    const uint4* src = (const uint4*)whh01;
    uint4* dst = (uint4*)whh_lds;
    for (int i = tid; i < (int)(2 * GRU_WMAT_H * 2 / 16); i += 192) dst[i] = src[i];
  }
  // init f16 h buffers from h0
  for (int i = tid; i < 16 * H_SZ; i += 192) {
    int r = i / H_SZ, c = i % H_SZ;
    hbuf[0][0][r][c] = (_Float16)h0g[(size_t)(m0 + r) * H_SZ + c];
    hbuf[1][0][r][c] = (_Float16)h0g[(size_t)B_SZ * H_SZ + (size_t)(m0 + r) * H_SZ + c];
  }
  __syncthreads();

  // persistent per-lane state (C-layout: element r -> row m0+rowOff+r, col N)
  float h0s[8], h1s[8];
  int   len[8];
#pragma unroll
  for (int r = 0; r < 8; ++r) {
    int row = m0 + rowOff + r;
    h0s[r] = h0g[(size_t)row * H_SZ + N];
    h1s[r] = h0g[(size_t)B_SZ * H_SZ + (size_t)row * H_SZ + N];
    len[r] = lengths[row];
  }
  const float bir0 = b_ih0[N],       bhr0 = b_hh0[N];
  const float biz0 = b_ih0[96 + N],  bhz0 = b_hh0[96 + N];
  const float bin0 = b_ih0[192 + N], bhn0 = b_hh0[192 + N];
  const float bir1 = b_ih1[N],       bhr1 = b_hh1[N];
  const float biz1 = b_ih1[96 + N],  bhz1 = b_hh1[96 + N];
  const float bin1 = b_ih1[192 + N], bhn1 = b_hh1[192 + N];

  // e_t tile: 16 contiguous rows x 192 B = 3072 B = 192 x uint4 (one per thread)
  auto eaddr = [&](int t) -> const uint4* {
    return (const uint4*)(e + ((size_t)t * B_SZ + m0) * H_SZ);
  };
  uint4 ereg = eaddr(0)[tid];            // prologue load for t=0

  int cur = 0;
  for (int t = 0; t < T_LEN; ++t) {
    ((uint4*)elds)[tid] = ereg;          // stage prefetched e_t into LDS
    __syncthreads();                     // e ready; prev-step h writes visible
    {                                    // prefetch e_{t+1} under this step's compute
      int tn = (t + 1 < T_LEN) ? (t + 1) : t;
      ereg = eaddr(tn)[tid];
    }
    const int nxt = cur ^ 1;

    gru_cell(elds, &hbuf[0][cur][0][0], wih0, whh_lds,
             wave, lane, bir0, bhr0, biz0, bhz0, bin0, bhn0,
             h0s, len, t, &hbuf[0][nxt][0][0]);
    __syncthreads();                     // layer-0 output ready for layer 1

    gru_cell(&hbuf[0][nxt][0][0], &hbuf[1][cur][0][0], wih1, whh_lds + GRU_WMAT_H,
             wave, lane, bir1, bhr1, biz1, bhz1, bin1, bhn1,
             h1s, len, t, &hbuf[1][nxt][0][0]);
    cur = nxt;                           // next iteration's first barrier covers h1 writes
  }

  // final hidden states straight into d_out's h section [2,B,96]
#pragma unroll
  for (int r = 0; r < 8; ++r) {
    int row = m0 + rowOff + r;
    hout[(size_t)row * H_SZ + N] = h0s[r];
    hout[(size_t)B_SZ * H_SZ + (size_t)row * H_SZ + N] = h1s[r];
  }
}

__global__ void head_kernel(const float* __restrict__ h2, const float* __restrict__ fc_w,
                            const float* __restrict__ fc_b, float* __restrict__ out) {
  int i = blockIdx.x * blockDim.x + threadIdx.x;     // b*8 + j
  if (i >= B_SZ * NACT) return;
  int b = i >> 3, j = i & 7;
  float acc = fc_b[j];
  for (int k = 0; k < H_SZ; ++k) {
    float h = h2[(size_t)b * H_SZ + k];
    h = (h > 0.0f) ? h : 0.01f * h;
    acc += h * fc_w[j * H_SZ + k];
  }
  out[i] = acc;
}

extern "C" void kernel_launch(void* const* d_in, const int* in_sizes, int n_in,
                              void* d_out, int out_size, void* d_ws, size_t ws_size,
                              hipStream_t stream) {
  const float* x      = (const float*)d_in[0];
  const int*   lens   = (const int*)  d_in[1];
  const float* h0     = (const float*)d_in[2];
  const float* emb_w1 = (const float*)d_in[3];
  const float* emb_b1 = (const float*)d_in[4];
  const float* emb_w2 = (const float*)d_in[5];
  const float* emb_b2 = (const float*)d_in[6];
  const float* w_ih0  = (const float*)d_in[7];
  const float* w_hh0  = (const float*)d_in[8];
  const float* b_ih0  = (const float*)d_in[9];
  const float* b_hh0  = (const float*)d_in[10];
  const float* w_ih1  = (const float*)d_in[11];
  const float* w_hh1  = (const float*)d_in[12];
  const float* b_ih1  = (const float*)d_in[13];
  const float* b_hh1  = (const float*)d_in[14];
  const float* fc_w   = (const float*)d_in[15];
  const float* fc_b   = (const float*)d_in[16];

  char* ws = (char*)d_ws;
  _Float16* w1p   = (_Float16*)(ws + OFF_W1P);
  _Float16* w2p   = (_Float16*)(ws + OFF_W2P);
  _Float16* wih0p = (_Float16*)(ws + OFF_WIH0);
  _Float16* wih1p = (_Float16*)(ws + OFF_WIH1);
  _Float16* whh0p = (_Float16*)(ws + OFF_WHH0);
  _Float16* whh1p = (_Float16*)(ws + OFF_WHH1);
  _Float16* e     = (_Float16*)(ws + OFF_E);

  // 1) pack weights to f16 WMMA B-fragments
  auto pack = [&](const float* w, _Float16* dst, int Kdim, int Nrows, int Kch, int Nt) {
    int total = Kch * Nt * 512;
    pack_b_kernel<<<(total + 255) / 256, 256, 0, stream>>>(w, dst, Kdim, Nrows, Nt, total);
  };
  pack(emb_w1, w1p,   DIN, E1,   12, 8);
  pack(emb_w2, w2p,   E1,  H_SZ,  4, 6);
  pack(w_ih0,  wih0p, H_SZ, 3 * H_SZ, 3, 18);
  pack(w_ih1,  wih1p, H_SZ, 3 * H_SZ, 3, 18);
  pack(w_hh0,  whh0p, H_SZ, 3 * H_SZ, 3, 18);
  pack(w_hh1,  whh1p, H_SZ, 3 * H_SZ, 3, 18);

  // 2) embedding MLP: 524288 rows -> 32768 row-tiles, 8 waves/block
  embed_kernel<<<4096, 256, 0, stream>>>(x, w1p, emb_b1, w2p, emb_b2, e);

  // 3) persistent fused 2-layer GRU (single launch; 128 blocks x 6 waves)
  float* out_ptr = (float*)d_out;                  // [B,8]
  float* h_ptr   = out_ptr + (size_t)B_SZ * NACT;  // [2,B,96]
  gru_fused_kernel<<<128, 192, 0, stream>>>(
      e, wih0p, wih1p, whh0p,
      b_ih0, b_hh0, b_ih1, b_hh1,
      h0, lens, h_ptr);

  // 4) head: out = LeakyReLU(h2) @ fc_w^T + fc_b  (reads h2 from d_out h section)
  const float* h2_final = h_ptr + (size_t)B_SZ * H_SZ;
  head_kernel<<<64, 256, 0, stream>>>(h2_final, fc_w, fc_b, out_ptr);
}
